// Brep_Gcn_54150947668368
// MI455X (gfx1250) — matrix operand order repaired and checked by reading, not compiled, not run on hardware
//
#include <hip/hip_runtime.h>

// ---------------------------------------------------------------- constants
#define NN        100000   // nodes (divisible by 16)
#define D_IN      83
#define K_PAD     96       // 83 padded to 3 * 32 for wmma K-steps
#define D_HID     1024
#define N_CLASSES 25
#define NC_PAD    32

typedef _Float16 half8 __attribute__((ext_vector_type(8)));
typedef _Float16 v16h  __attribute__((ext_vector_type(16)));
typedef float    v8f   __attribute__((ext_vector_type(8)));

// ------------------------------------------------- WMMA fragment loaders
// A-matrix 16x32 f16 layout (ISA 7.12.2): lanes 0-15 hold row M=lane with
// K = {k0..k0+7, k0+16..k0+23}; lanes 16-31 hold row M=lane-16 with
// K = {k0+8..k0+15, k0+24..k0+31}.  src is row-major [rows][ld].
__device__ __forceinline__ v16h load_frag_a(const _Float16* __restrict__ base,
                                            int ld, int m0, int k0, int lane) {
  int l  = lane & 15;
  int hs = lane >> 4;
  const _Float16* p = base + (size_t)(m0 + l) * ld + k0 + hs * 8;
  half8 lo = *(const half8*)p;          // K block 0
  half8 hi = *(const half8*)(p + 16);   // K block +16
  return __builtin_shufflevector(lo, hi, 0,1,2,3,4,5,6,7,8,9,10,11,12,13,14,15);
}

// B-matrix 32x16 f16 (K x N): lanes 0-15 hold col N=lane with K={k0..k0+7, k0+16..k0+23},
// lanes 16-31 hold col N=lane-16 with K={k0+8..k0+15, k0+24..k0+31}.  src is packed
// column-major: Bc[n][k] with leading dim ld (contiguous K per column).
__device__ __forceinline__ v16h load_frag_b(const _Float16* __restrict__ base,
                                            int ld, int n0, int k0, int lane) {
  int l  = lane & 15;
  int hs = lane >> 4;
  const _Float16* p = base + (size_t)(n0 + l) * ld + k0 + hs * 8;
  half8 lo = *(const half8*)p;
  half8 hi = *(const half8*)(p + 16);
  return __builtin_shufflevector(lo, hi, 0,1,2,3,4,5,6,7,8,9,10,11,12,13,14,15);
}

// ------------------------------------------------------------ prep kernels
__global__ void k_wsum(const float* __restrict__ cw, const float* __restrict__ cb,
                       float* __restrict__ wsum) {
  if (blockIdx.x == 0 && threadIdx.x == 0) {
    for (int k = 0; k < 5; ++k)
      wsum[k] = cw[k] + cw[5 + k] + cw[10 + k] + cw[15 + k];
    wsum[5] = cb[0] + cb[1] + cb[2] + cb[3];
  }
}

// Conv1d(1->4,k5,pad2), sum channels, relu; write f16 padded rows [NN][K_PAD]
__global__ void k_conv(const float* __restrict__ feat, const float* __restrict__ wsum,
                       _Float16* __restrict__ Xh, int total) {
  int idx = blockIdx.x * blockDim.x + threadIdx.x;
  if (idx >= total) return;
  int i = idx % K_PAD, n = idx / K_PAD;
  float r = 0.f;
  if (i < D_IN) {
    float acc = wsum[5];
#pragma unroll
    for (int k = 0; k < 5; ++k) {
      int j = i + k - 2;
      if (j >= 0 && j < D_IN) acc += wsum[k] * feat[(size_t)n * D_IN + j];
    }
    r = fmaxf(acc, 0.f);
  }
  Xh[idx] = (_Float16)r;
}

// W1 [83,1024] f32 -> column-major f16 W1c[n][k], K padded to 96
__global__ void k_packW1(const float* __restrict__ W1, _Float16* __restrict__ W1c) {
  int idx = blockIdx.x * blockDim.x + threadIdx.x;
  if (idx >= D_HID * K_PAD) return;
  int n = idx / K_PAD, k = idx % K_PAD;
  W1c[idx] = (_Float16)(k < D_IN ? W1[(size_t)k * D_HID + n] : 0.f);
}

// W2 [1024,25] f32 -> column-major f16 W2c[n][k], N padded to 32
__global__ void k_packW2(const float* __restrict__ W2, _Float16* __restrict__ W2c) {
  int idx = blockIdx.x * blockDim.x + threadIdx.x;
  if (idx >= NC_PAD * D_HID) return;
  int n = idx / D_HID, k = idx % D_HID;
  W2c[idx] = (_Float16)(n < N_CLASSES ? W2[(size_t)k * N_CLASSES + n] : 0.f);
}

__global__ void k_zero(float4* __restrict__ p, int n4) {
  int idx = blockIdx.x * blockDim.x + threadIdx.x;
  if (idx < n4) p[idx] = make_float4(0.f, 0.f, 0.f, 0.f);
}

__global__ void k_init_out(const float* __restrict__ b2, float* __restrict__ out, int total) {
  int idx = blockIdx.x * blockDim.x + threadIdx.x;
  if (idx < total) out[idx] = b2[idx % N_CLASSES];
}

// ------------------------------------------------------------- GEMM 1
// Yh[NN][1024] f16 = Xh[NN][96] @ W1c^T ; wave tile 16x64, block 128 thr.
__global__ void k_gemm1(const _Float16* __restrict__ Xh,
                        const _Float16* __restrict__ W1c,
                        _Float16* __restrict__ Yh) {
  int lane  = threadIdx.x & 31;
  int wave  = threadIdx.x >> 5;                 // 0..3
  int m0    = blockIdx.x * 16;                  // gridDim.x = NN/16
  int nbase = blockIdx.y * 256 + wave * 64;     // gridDim.y = 4
  v8f acc[4] = {v8f{}, v8f{}, v8f{}, v8f{}};
#pragma unroll
  for (int k0 = 0; k0 < K_PAD; k0 += 32) {
    v16h a = load_frag_a(Xh, K_PAD, m0, k0, lane);
#pragma unroll
    for (int i = 0; i < 4; ++i) {
      v16h b = load_frag_b(W1c, K_PAD, nbase + 16 * i, k0, lane);
      acc[i] = __builtin_amdgcn_wmma_f32_16x16x32_f16(
          false, a, false, b, (short)0, acc[i], false, false);
    }
  }
  // C/D layout: VGPR v -> row m0+v (lanes 0-15) / m0+v+8 (lanes 16-31), col = N lane
  int l = lane & 15, hs = lane >> 4;
#pragma unroll
  for (int i = 0; i < 4; ++i) {
    int col = nbase + 16 * i + l;
#pragma unroll
    for (int v = 0; v < 8; ++v) {
      int row = m0 + v + 8 * hs;
      Yh[(size_t)row * D_HID + col] = (_Float16)acc[i][v];
    }
  }
}

// ------------------------------------------------------------- SPMM 1
// Hacc[row] += val * Yh[col]; one block per edge, 128 thr * 8 halves = 1024.
__global__ void k_spmm1(const _Float16* __restrict__ Yh,
                        const float* __restrict__ val,
                        const int* __restrict__ row,
                        const int* __restrict__ col,
                        float* __restrict__ Hacc, int E) {
  int e = blockIdx.x;
  int t = threadIdx.x;
  int c = col[e], r = row[e];
  float v = val[e];
  int ef = e + 96;                       // prefetch a future edge's gather row
  if (ef < E) {
    int cf = col[ef];
    __builtin_prefetch(Yh + (size_t)cf * D_HID + t * 8, 0, 0);
  }
  const half8 y = *(const half8*)(Yh + (size_t)c * D_HID + t * 8);
  float* dst = Hacc + (size_t)r * D_HID + t * 8;
#pragma unroll
  for (int j = 0; j < 8; ++j) atomicAdd(dst + j, v * (float)y[j]);
}

// relu(Hacc + b1) -> f16 Hh (aliased onto Yh's workspace region)
__global__ void k_relu_b1(const float* __restrict__ Hacc, const float* __restrict__ b1,
                          _Float16* __restrict__ Hh, int total) {
  int idx = blockIdx.x * blockDim.x + threadIdx.x;
  if (idx >= total) return;
  float x = Hacc[idx] + b1[idx & (D_HID - 1)];
  Hh[idx] = (_Float16)fmaxf(x, 0.f);
}

// ------------------------------------------------------------- GEMM 2
// Z[NN][32] f32 = Hh[NN][1024] @ W2c^T ; wave tile 16x32, block 128 thr.
__global__ void k_gemm2(const _Float16* __restrict__ Hh,
                        const _Float16* __restrict__ W2c,
                        float* __restrict__ Z) {
  int lane = threadIdx.x & 31;
  int wave = threadIdx.x >> 5;
  int m0   = (blockIdx.x * 4 + wave) * 16;
  if (m0 >= NN) return;                      // wave-uniform guard: EXEC stays full
  v8f acc[2] = {v8f{}, v8f{}};
  for (int k0 = 0; k0 < D_HID; k0 += 32) {
    v16h a = load_frag_a(Hh, D_HID, m0, k0, lane);
#pragma unroll
    for (int i = 0; i < 2; ++i) {
      v16h b = load_frag_b(W2c, D_HID, 16 * i, k0, lane);
      acc[i] = __builtin_amdgcn_wmma_f32_16x16x32_f16(
          false, a, false, b, (short)0, acc[i], false, false);
    }
  }
  int l = lane & 15, hs = lane >> 4;
#pragma unroll
  for (int i = 0; i < 2; ++i) {
    int colz = 16 * i + l;
#pragma unroll
    for (int v = 0; v < 8; ++v) {
      int rowz = m0 + v + 8 * hs;
      Z[(size_t)rowz * NC_PAD + colz] = acc[i][v];
    }
  }
}

// ------------------------------------------------------------- SPMM 2
// out[row, c] += val * Z[col, c]; 32 lanes per edge (25 active), 8 edges/block.
__global__ void k_spmm2(const float* __restrict__ Z,
                        const float* __restrict__ val,
                        const int* __restrict__ row,
                        const int* __restrict__ col,
                        float* __restrict__ out, int E) {
  int lane = threadIdx.x & 31;
  int e = blockIdx.x * 8 + (threadIdx.x >> 5);
  if (e >= E) return;
  int c = col[e], r = row[e];
  float v = val[e];
  int ef = e + 512;
  if (ef < E && lane == 0) __builtin_prefetch(Z + (size_t)col[ef] * NC_PAD, 0, 0);
  if (lane < N_CLASSES)
    atomicAdd(out + (size_t)r * N_CLASSES + lane, v * Z[(size_t)c * NC_PAD + lane]);
}

// ------------------------------------------------------------- launcher
extern "C" void kernel_launch(void* const* d_in, const int* in_sizes, int n_in,
                              void* d_out, int out_size, void* d_ws, size_t ws_size,
                              hipStream_t stream) {
  const float* feature = (const float*)d_in[0];
  const float* conv_w  = (const float*)d_in[1];
  const float* conv_b  = (const float*)d_in[2];
  const float* W1      = (const float*)d_in[3];
  const float* b1      = (const float*)d_in[4];
  const float* W2      = (const float*)d_in[5];
  const float* b2      = (const float*)d_in[6];
  const float* adj_val = (const float*)d_in[7];
  const int*   e_row   = (const int*)d_in[8];
  const int*   e_col   = (const int*)d_in[9];
  float* out = (float*)d_out;
  const int E = in_sizes[7];

  // workspace carve-up (256B aligned)
  char* ws = (char*)d_ws;
  size_t off = 0;
  auto carve = [&](size_t bytes) -> char* {
    off = (off + 255) & ~(size_t)255;
    char* p = ws + off;
    off += bytes;
    return p;
  };
  float*    wsum = (float*)   carve(8 * sizeof(float));
  _Float16* Xh   = (_Float16*)carve((size_t)NN * K_PAD * 2);
  _Float16* W1c  = (_Float16*)carve((size_t)D_HID * K_PAD * 2);
  _Float16* W2c  = (_Float16*)carve((size_t)NC_PAD * D_HID * 2);
  _Float16* Yh   = (_Float16*)carve((size_t)NN * D_HID * 2);
  float*    Hacc = (float*)   carve((size_t)NN * D_HID * 4);
  float*    Z    = (float*)   carve((size_t)NN * NC_PAD * 4);
  _Float16* Hh   = Yh;  // Yh dead after spmm1; reuse for relu output

  // 1) collapse conv channels, 2) prelude, 3-4) weight repack
  k_wsum<<<1, 32, 0, stream>>>(conv_w, conv_b, wsum);
  {
    int total = NN * K_PAD;
    k_conv<<<(total + 255) / 256, 256, 0, stream>>>(feature, wsum, Xh, total);
  }
  k_packW1<<<(D_HID * K_PAD + 255) / 256, 256, 0, stream>>>(W1, W1c);
  k_packW2<<<(NC_PAD * D_HID + 255) / 256, 256, 0, stream>>>(W2, W2c);

  // 5) GEMM1 via WMMA
  k_gemm1<<<dim3(NN / 16, 4), 128, 0, stream>>>(Xh, W1c, Yh);

  // 6-7) zero accumulator, scatter-gather SPMM
  {
    int n4 = NN * D_HID / 4;
    k_zero<<<(n4 + 255) / 256, 256, 0, stream>>>((float4*)Hacc, n4);
  }
  k_spmm1<<<E, 128, 0, stream>>>(Yh, adj_val, e_row, e_col, Hacc, E);

  // 8) relu + bias -> f16
  {
    int total = NN * D_HID;
    k_relu_b1<<<(total + 255) / 256, 256, 0, stream>>>(Hacc, b1, Hh, total);
  }

  // 9) GEMM2 via WMMA
  k_gemm2<<<(NN / 16 + 3) / 4, 128, 0, stream>>>(Hh, W2c, Z);

  // 10-11) bias-init output, SPMM2
  {
    int total = NN * N_CLASSES;
    k_init_out<<<(total + 255) / 256, 256, 0, stream>>>(b2, out, total);
  }
  k_spmm2<<<(E + 7) / 8, 256, 0, stream>>>(Z, adj_val, e_row, e_col, out, E);
}
